// GraphEncoder_12575664243381
// MI455X (gfx1250) — compile-verified, matrix operand order checked by reading, not run on hardware
//
#include <hip/hip_runtime.h>

// ---------- types for WMMA ----------
typedef __attribute__((ext_vector_type(16))) __bf16 v16bf;
typedef __attribute__((ext_vector_type(8)))  float  v8f;

union Frag {
    v16bf v;
    uint4 q[2];      // 2 x 16B = 32B = 16 bf16 per lane
};

// float -> bf16 (round to nearest even), as raw ushort
__device__ __forceinline__ unsigned short f2bf(float f) {
    unsigned int u = __float_as_uint(f);
    u += 0x7FFFu + ((u >> 16) & 1u);
    return (unsigned short)(u >> 16);
}

// ---------- degree / norm ----------
__global__ void k_fill1(float* p, int n) {
    int i = blockIdx.x * blockDim.x + threadIdx.x;
    if (i < n) p[i] = 1.0f;                 // self-loop contributes 1 to degree
}

__global__ void k_degree(const long long* __restrict__ dst, float* deg, int e) {
    int i = blockIdx.x * blockDim.x + threadIdx.x;
    if (i < e) atomicAdd(&deg[(int)dst[i]], 1.0f);
}

__global__ void k_rsqrt_inplace(float* p, int n) {
    int i = blockIdx.x * blockDim.x + threadIdx.x;
    if (i < n) p[i] = rsqrtf(p[i]);         // deg >= 1 always (self loop)
}

// ---------- conversions ----------
__global__ void k_f2bf(const float* __restrict__ in, unsigned short* __restrict__ out, int n) {
    int i = blockIdx.x * blockDim.x + threadIdx.x;
    if (i < n) out[i] = f2bf(in[i]);
}

__global__ void k_relu_f2bf(const float* __restrict__ in, unsigned short* __restrict__ out, int n) {
    int i = blockIdx.x * blockDim.x + threadIdx.x;
    if (i < n) out[i] = f2bf(fmaxf(in[i], 0.0f));
}

// ---------- pack W[128x128] f32 row-major -> bf16 WMMA B-fragment layout ----------
// layout: Wp[tile=kt*8+nt][lane 0..31][e 0..15], so each lane's 16 bf16 are contiguous (32B).
// B 32x16 bf16 layout (ISA 7.12.2): lane = {half,n}, VGPR j<4 -> K = half*8+2j(+1),
//                                   VGPR j>=4 -> K = 16+half*8+2(j-4)(+1).
__global__ void k_packW(const float* __restrict__ W, unsigned short* __restrict__ Wp) {
    int idx = blockIdx.x * blockDim.x + threadIdx.x;     // 0 .. 4*8*32*16-1
    if (idx >= 4 * 8 * 32 * 16) return;
    int e    = idx & 15;
    int lane = (idx >> 4) & 31;
    int tile = idx >> 9;
    int kt = tile >> 3, nt = tile & 7;
    int half = lane >> 4;
    int n = nt * 16 + (lane & 15);
    int j = e >> 1;
    int kk = ((j < 4) ? (half * 8 + 2 * j) : (16 + half * 8 + 2 * (j - 4))) + (e & 1);
    int k = kt * 32 + kk;
    Wp[idx] = f2bf(W[k * 128 + n]);
}

// ---------- WMMA GEMM: Out[rows,128] f32 = A[rows,128] bf16 @ W[128,128] (packed) ----------
// One wave per 16-row slab; 8 waves per block. 4 K-steps x 8 N-tiles = 32 v_wmma per wave.
__global__ void __launch_bounds__(256)
k_gemm_bf16(const unsigned short* __restrict__ A,
            const unsigned short* __restrict__ Wp,
            float* __restrict__ Out, int rows) {
    int wave = (int)(blockIdx.x * 8 + (threadIdx.x >> 5));
    int lane = threadIdx.x & 31;
    int rowBase = wave * 16;
    if (rowBase >= rows) return;                         // wave-uniform; EXEC stays all-ones

    int half = lane >> 4;
    int mr   = lane & 15;
    const unsigned short* arow = A + (size_t)(rowBase + mr) * 128;

    v8f acc[8];
#pragma unroll
    for (int nt = 0; nt < 8; ++nt) acc[nt] = (v8f)(0.0f);

#pragma unroll
    for (int kt = 0; kt < 4; ++kt) {
        Frag a;
        // A 16x32 bf16 fragment: lane holds K[half*8 .. +7] and K[16+half*8 .. +7]
        a.q[0] = *(const uint4*)(arow + kt * 32 + half * 8);
        a.q[1] = *(const uint4*)(arow + kt * 32 + 16 + half * 8);
#pragma unroll
        for (int nt = 0; nt < 8; ++nt) {
            Frag b;
            const uint4* bp = (const uint4*)(Wp + ((size_t)(kt * 8 + nt) * 32 + lane) * 16);
            b.q[0] = bp[0];
            b.q[1] = bp[1];
            acc[nt] = __builtin_amdgcn_wmma_f32_16x16x32_bf16(
                false, a.v, false, b.v, (short)0, acc[nt], false, false);
        }
    }

    // D 16x16 f32 layout: VGPR r -> M = half*8 + r, N = nt*16 + (lane&15)
#pragma unroll
    for (int nt = 0; nt < 8; ++nt) {
#pragma unroll
        for (int r = 0; r < 8; ++r) {
            Out[(size_t)(rowBase + half * 8 + r) * 128 + nt * 16 + (lane & 15)] = acc[nt][r];
        }
    }
}

// ---------- aggregation ----------
// agg[v,c] = h[v,c]*dinv[v]^2 + bias[c]   (self-loop contribution + bias)
__global__ void k_self_init(const float* __restrict__ h, const float* __restrict__ dinv,
                            const float* __restrict__ bias, float* __restrict__ agg, int n) {
    int i = blockIdx.x * blockDim.x + threadIdx.x;
    if (i >= n) return;
    int v = i >> 7;
    int c = i & 127;
    float dv = dinv[v];
    agg[i] = h[i] * dv * dv + bias[c];
}

// One wave per edge; lane l handles channels [4l, 4l+4): coalesced float4 gather of h[src],
// four f32 atomic adds into agg[dst] (L2-resident, both buffers fit in 192MB L2).
__global__ void __launch_bounds__(256)
k_edge_agg(const long long* __restrict__ src, const long long* __restrict__ dst,
           const float* __restrict__ dinv, const float* __restrict__ h,
           float* __restrict__ agg, int e) {
    int wave = (int)((blockIdx.x * blockDim.x + threadIdx.x) >> 5);
    int lane = threadIdx.x & 31;
    if (wave >= e) return;
    int s = (int)src[wave];
    int d = (int)dst[wave];
    float norm = dinv[s] * dinv[d];
    float4 v = ((const float4*)(h + (size_t)s * 128))[lane];
    float* ad = agg + (size_t)d * 128 + lane * 4;
    atomicAdd(ad + 0, v.x * norm);
    atomicAdd(ad + 1, v.y * norm);
    atomicAdd(ad + 2, v.z * norm);
    atomicAdd(ad + 3, v.w * norm);
}

// ---------- host launch ----------
extern "C" void kernel_launch(void* const* d_in, const int* in_sizes, int n_in,
                              void* d_out, int out_size, void* d_ws, size_t ws_size,
                              hipStream_t stream) {
    const float*      x  = (const float*)d_in[0];
    const long long*  ei = (const long long*)d_in[1];   // int64 edge_index [2,E]
    const float*      W1 = (const float*)d_in[2];
    const float*      b1 = (const float*)d_in[3];
    const float*      W2 = (const float*)d_in[4];
    const float*      b2 = (const float*)d_in[5];

    const int N = in_sizes[0] / 128;
    const int E = in_sizes[1] / 2;
    const long long* src = ei;
    const long long* dst = ei + E;

    const size_t NC   = (size_t)N * 128;
    const int    PACKN = 4 * 8 * 32 * 16;               // 16384 bf16 per packed W

    // workspace layout (256B aligned)
    char* ws = (char*)d_ws;
    size_t off = 0;
    auto take = [&](size_t bytes) { char* p = ws + off; off = (off + bytes + 255) & ~(size_t)255; return p; };
    float*          dinv = (float*)take((size_t)N * 4);
    unsigned short* W1p  = (unsigned short*)take((size_t)PACKN * 2);
    unsigned short* W2p  = (unsigned short*)take((size_t)PACKN * 2);
    unsigned short* xb   = (unsigned short*)take(NC * 2);   // reused as h1 bf16
    float*          hlin = (float*)take(NC * 4);            // GEMM output (both layers)
    float*          agg  = (float*)take(NC * 4);            // layer-1 aggregation

    float* out = (float*)d_out;

    const int T = 256;
    dim3 blk(T);
    int gN    = (N + T - 1) / T;
    int gE    = (E + T - 1) / T;
    int gNC   = (int)((NC + T - 1) / T);
    int gPack = (PACKN + T - 1) / T;
    int gGemm = ((N + 15) / 16 + 7) / 8;                 // 16 rows/wave, 8 waves/block
    int gEdge = (E + 7) / 8;                              // 1 wave/edge, 8 edges/block

    // degrees -> dinv
    k_fill1<<<gN, blk, 0, stream>>>(dinv, N);
    k_degree<<<gE, blk, 0, stream>>>(dst, dinv, E);
    k_rsqrt_inplace<<<gN, blk, 0, stream>>>(dinv, N);

    // precision staging
    k_f2bf<<<gNC, blk, 0, stream>>>(x, xb, (int)NC);
    k_packW<<<gPack, blk, 0, stream>>>(W1, W1p);
    k_packW<<<gPack, blk, 0, stream>>>(W2, W2p);

    // ----- layer 1 -----
    k_gemm_bf16<<<gGemm, blk, 0, stream>>>(xb, W1p, hlin, N);
    k_self_init<<<gNC, blk, 0, stream>>>(hlin, dinv, b1, agg, (int)NC);
    k_edge_agg<<<gEdge, blk, 0, stream>>>(src, dst, dinv, hlin, agg, E);
    k_relu_f2bf<<<gNC, blk, 0, stream>>>(agg, xb, (int)NC);   // ReLU + bf16, reuse xb

    // ----- layer 2 -----
    k_gemm_bf16<<<gGemm, blk, 0, stream>>>(xb, W2p, hlin, N);
    k_self_init<<<gNC, blk, 0, stream>>>(hlin, dinv, b2, out, (int)NC);
    k_edge_agg<<<gEdge, blk, 0, stream>>>(src, dst, dinv, hlin, out, E);
}
